// VAErec_36369783063088
// MI455X (gfx1250) — matrix operand order repaired and checked by reading, not compiled, not run on hardware
//
#include <hip/hip_runtime.h>

// ---------------------------------------------------------------------------
// CDNA5 WMMA (wave32): D(f32 16x16) = A(bf16 16x32) * B(bf16 32x16) + C
// ---------------------------------------------------------------------------
typedef __bf16 v16bf __attribute__((ext_vector_type(16)));
typedef __bf16 v8bf  __attribute__((ext_vector_type(8)));
typedef __bf16 v4bf  __attribute__((ext_vector_type(4)));
typedef float  v8f   __attribute__((ext_vector_type(8)));

#define DEV __device__ __forceinline__

DEV v8f wmma_bf16(v16bf a, v16bf b, v8f c) {
  return __builtin_amdgcn_wmma_f32_16x16x32_bf16(false, a, false, b, (short)0, c,
                                                 false, false);
}

// Fast branch-free transcendentals on native v_exp_f32 / v_rcp_f32.
DEV float fast_sigmoid(float x) {
  return __builtin_amdgcn_rcpf(1.0f + __builtin_amdgcn_exp2f(x * -1.442695041f));
}
DEV float fast_tanh(float x) {
  const float xc = fminf(fmaxf(x, -10.0f), 10.0f);
  const float e = __builtin_amdgcn_exp2f(xc * 2.885390082f);  // exp(2x)
  return (e - 1.0f) * __builtin_amdgcn_rcpf(e + 1.0f);
}

// A fragment (16x32 bf16) from a bf16 row-major LDS buffer (zero-padded, LDA
// multiple of 8 -> both 16B segments aligned; no conversion, no guards).
// ISA 16-bit A layout: lane m=lane&15, half=lane>>4; elems 0..7 = K in
// [kt*32+8*half, +8), elems 8..15 = K in [kt*32+16+8*half, +8).
template <int LDA>
DEV v16bf load_A_bf(const __bf16* A, int kt, int lane) {
  const int m = lane & 15, half = lane >> 4;
  const __bf16* rp = A + m * LDA + kt * 32 + 8 * half;
  const v8bf lo = *(const v8bf*)rp;
  const v8bf hi = *(const v8bf*)(rp + 16);
  v16bf f;
#pragma unroll
  for (int i = 0; i < 8; ++i) {
    f[i]     = lo[i];
    f[i + 8] = hi[i];
  }
  return f;
}

// B fragment: pre-packed bf16 weights, one contiguous 32B chunk per lane.
DEV v16bf load_B_frag(const unsigned short* __restrict__ Wp, int KT, int nt,
                      int kt, int lane) {
  return *reinterpret_cast<const v16bf*>(
      Wp + ((size_t)((nt * KT + kt) * 32 + lane)) * 16);
}

// Tile GEMM accumulating into acc, software-pipelined B loads, bf16 LDS A.
template <int KT, int LDA>
DEV v8f gemm_bf(const __bf16* A, const unsigned short* __restrict__ Wp, int nt,
                int lane, v8f acc) {
  v16bf b = load_B_frag(Wp, KT, nt, 0, lane);
#pragma unroll
  for (int kt = 0; kt < KT; ++kt) {
    v16bf bn = (kt + 1 < KT) ? load_B_frag(Wp, KT, nt, kt + 1, lane) : b;
    v16bf a = load_A_bf<LDA>(A, kt, lane);
    acc = wmma_bf16(a, b, acc);
    b = bn;
  }
  return acc;
}

// ---------------------------------------------------------------------------
// GRU GEMM phase. U (G view): Grz[16][256] = (i_r+h_r | i_z+h_z) with biases,
// Gxn[16][128] = i_n + b, Ghn[16][128] = h_n + b.  The r/z tiles chain the
// x-GEMM and h-GEMM through the WMMA C operand (one store of the sum).
// ---------------------------------------------------------------------------
template <int KTX, int LDAX>
DEV void gru_phase(const __bf16* AxB, const __bf16* hB, float* U,
                   const unsigned short* __restrict__ WihP,
                   const unsigned short* __restrict__ WhhP,
                   const float* __restrict__ b_ih,
                   const float* __restrict__ b_hh, int wave, int lane) {
  const int half = lane >> 4, nl = lane & 15;
  const v8f vz = {};
#pragma unroll
  for (int s = 0; s < 2; ++s) {  // s=0: r gates (nt 0..7), s=1: z gates (8..15)
    const int nt = s * 8 + wave;
    v8f acc = gemm_bf<KTX, LDAX>(AxB, WihP, nt, lane, vz);
    acc = gemm_bf<4, 128>(hB, WhhP, nt, lane, acc);
    const int col = nt * 16 + nl;
    const float bias = b_ih[col] + b_hh[col];
#pragma unroll
    for (int v = 0; v < 8; ++v) U[(v + 8 * half) * 256 + col] = acc[v] + bias;
  }
  {  // n gates (nt 16..23): x and h parts kept separate
    const int nt = 16 + wave;
    const int colN = wave * 16 + nl;
    v8f ax = gemm_bf<KTX, LDAX>(AxB, WihP, nt, lane, vz);
    v8f ah = gemm_bf<4, 128>(hB, WhhP, nt, lane, vz);
    const float bx = b_ih[256 + colN], bh = b_hh[256 + colN];
#pragma unroll
    for (int v = 0; v < 8; ++v) {
      U[4096 + (v + 8 * half) * 128 + colN] = ax[v] + bx;
      U[6144 + (v + 8 * half) * 128 + colN] = ah[v] + bh;
    }
  }
}

// Vectorized gate math: 4 consecutive cols per thread; updates fp32 h state
// and the packed bf16 copy used as the WMMA A operand.
DEV void gate_phase(const float* U, float* hS, __bf16* hB, int tid) {
#pragma unroll
  for (int e = 0; e < 2; ++e) {
    const int t = tid + e * 256;   // 0..511
    const int r = t >> 5;          // 0..15
    const int j0 = (t & 31) << 2;  // 0,4,...,124
    const float4 rr = *(const float4*)(U + r * 256 + j0);
    const float4 zz = *(const float4*)(U + r * 256 + 128 + j0);
    const float4 xn = *(const float4*)(U + 4096 + r * 128 + j0);
    const float4 hn = *(const float4*)(U + 6144 + r * 128 + j0);
    float4* hp = (float4*)(hS + r * 128 + j0);
    float4 hv = *hp;
    float4 o;
#define GRU1(c)                                     \
    {                                               \
      const float rg = fast_sigmoid(rr.c);          \
      const float zg = fast_sigmoid(zz.c);          \
      const float ng = fast_tanh(xn.c + rg * hn.c); \
      o.c = (1.0f - zg) * ng + zg * hv.c;           \
    }
    GRU1(x) GRU1(y) GRU1(z) GRU1(w)
#undef GRU1
    *hp = o;
    v4bf ob;
    ob[0] = (__bf16)o.x; ob[1] = (__bf16)o.y;
    ob[2] = (__bf16)o.z; ob[3] = (__bf16)o.w;
    *(v4bf*)(hB + r * 128 + j0) = ob;
  }
}

// Two-layer tanh MLP, bf16 outputs in the U union:
// m1 = [16x320] bf16 (300 valid), m2 = m1+16*320 [16x224] bf16 (200 valid).
DEV void mlp2(const __bf16* hB, __bf16* mB,
              const unsigned short* __restrict__ W1p,
              const float* __restrict__ b1p,
              const unsigned short* __restrict__ W2p,
              const float* __restrict__ b2p, int tid, int wave, int lane) {
  __bf16* m1 = mB;
  __bf16* m2 = mB + 16 * 320;
  const int half = lane >> 4, nl = lane & 15;
  const v8f vz = {};
  {  // zero pad columns (re-done each call; U is clobbered every step)
    const int r = tid >> 4, c = tid & 15;  // 256 threads == 16x16
    m1[r * 320 + 304 + c] = (__bf16)0.0f;
    m2[r * 224 + 208 + c] = (__bf16)0.0f;
  }
  for (int nt = wave; nt < 19; nt += 8) {  // 128 -> 300(+pad 304), tanh
    v8f acc = gemm_bf<4, 128>(hB, W1p, nt, lane, vz);
    const int col = nt * 16 + nl;
    const float b = b1p[col];
#pragma unroll
    for (int v = 0; v < 8; ++v)
      m1[(v + 8 * half) * 320 + col] = (__bf16)fast_tanh(acc[v] + b);
  }
  __syncthreads();
  for (int nt = wave; nt < 13; nt += 8) {  // 300 -> 200(+pad 208), tanh
    v8f acc = gemm_bf<10, 320>(m1, W2p, nt, lane, vz);
    const int col = nt * 16 + nl;
    const float b = b2p[col];
#pragma unroll
    for (int v = 0; v < 8; ++v)
      m2[(v + 8 * half) * 224 + col] = (__bf16)fast_tanh(acc[v] + b);
  }
  __syncthreads();
}

// Stage a 16x144 fp32 global strip into a 16x160 bf16 LDS tile (pad pre-zeroed).
DEV void stage_x(const float* __restrict__ xt, __bf16* xB, int tid) {
  for (int q = tid; q < 576; q += 256) {  // 576 float4 quads
    const int r = q / 36, c4 = (q - r * 36) * 4;
    const float4 v = *(const float4*)(xt + r * 144 + c4);
    v4bf o;
    o[0] = (__bf16)v.x; o[1] = (__bf16)v.y;
    o[2] = (__bf16)v.z; o[3] = (__bf16)v.w;
    *(v4bf*)(xB + r * 160 + c4) = o;
  }
}

// ---------------------------------------------------------------------------
// Fused VAE kernel: each block owns a 16-row batch strip end-to-end.
// ---------------------------------------------------------------------------
struct Params {
  const float* x;
  float* xrec; float* mu; float* lv;
  const unsigned short *PwihE, *PwhhE, *PwihD, *PwhhD;
  const unsigned short *PW1e, *PW2e, *PWmu, *PWlv, *PW1d, *PW2d, *PWo;
  const float *bihE, *bhhE, *bihD, *bhhD;
  const float *b1ep, *b2ep, *b1dp, *b2dp, *bmu, *blv, *bo;
};

__global__ __launch_bounds__(256) void vae_all(Params p) {
  __shared__ float hS[16 * 128];                    // fp32 h state
  __shared__ __align__(16) __bf16 hB[16 * 128];     // bf16 h copy (A operand)
  __shared__ __align__(16) __bf16 xrnB[5120];       // enc: xB[2][16*160] / dec: rnnB[16*288]
  __shared__ __align__(16) float U[16 * 512];       // union: {Grz|Gxn|Ghn} f32 / {m1|m2} bf16
  const int tid = threadIdx.x;
  const int wave = tid >> 5, lane = tid & 31;
  const int half = lane >> 4, nl = lane & 15;
  const int row0 = blockIdx.x * 16;
  const v8f vz = {};

  // init: h0 = 0; zero the x-tile pad columns of both staging buffers
  for (int i = tid; i < 16 * 128; i += 256) { hS[i] = 0.0f; hB[i] = (__bf16)0.0f; }
  for (int i = tid; i < 512; i += 256) {
    const int bq = i >> 8, r = (i >> 4) & 15, c = i & 15;
    xrnB[bq * 2560 + r * 160 + 144 + c] = (__bf16)0.0f;
  }
  stage_x(p.x + (size_t)row0 * 144, xrnB, tid);  // t = 0
  __syncthreads();

  // ---------------- encoder: 90 GRU steps over time ----------------
#pragma unroll 1
  for (int t = 0; t < 90; ++t) {
    const __bf16* xB = xrnB + (t & 1) * 2560;
    gru_phase<5, 160>(xB, hB, U, p.PwihE, p.PwhhE, p.bihE, p.bhhE, wave, lane);
    if (t + 1 < 90)  // overlap next strip's staging with this step's GEMMs
      stage_x(p.x + (size_t)(t + 1) * (2048 * 144) + (size_t)row0 * 144,
              xrnB + ((t + 1) & 1) * 2560, tid);
    __syncthreads();
    gate_phase(U, hS, hB, tid);
    __syncthreads();
  }

  // ---------------- encoder MLP + latent heads ----------------
  __bf16* mB = (__bf16*)U;
  mlp2(hB, mB, p.PW1e, p.b1ep, p.PW2e, p.b2ep, tid, wave, lane);  // ends synced
  __bf16* rnnB = xrnB;  // decoder rnn_in strip: x(144) | z(128) | pad(16), bf16
  {  // heads: one 16-col tile per wave (8*16 = 128), shared A fragments
    const __bf16* m2 = mB + 16 * 320;
    v8f am = vz, al = vz;
#pragma unroll
    for (int kt = 0; kt < 7; ++kt) {
      v16bf a = load_A_bf<224>(m2, kt, lane);
      am = wmma_bf16(a, load_B_frag(p.PWmu, 7, wave, kt, lane), am);
      al = wmma_bf16(a, load_B_frag(p.PWlv, 7, wave, kt, lane), al);
    }
    const int col = wave * 16 + nl;  // 0..127
#pragma unroll
    for (int v = 0; v < 8; ++v) {
      const int rl = v + 8 * half;
      const float m = am[v] + p.bmu[col];
      p.mu[(size_t)(row0 + rl) * 128 + col] = m;
      rnnB[rl * 288 + 144 + col] = (__bf16)m;  // z = mu, constant across decode
      p.lv[(size_t)(row0 + rl) * 128 + col] = al[v] + p.blv[col];
    }
  }
  // decoder init: hd = 0, rnn x-part = x[0] strip (bf16), pad = 0
  for (int i = tid; i < 16 * 128; i += 256) { hS[i] = 0.0f; hB[i] = (__bf16)0.0f; }
  for (int q = tid; q < 576; q += 256) {
    const int r = q / 36, c4 = (q - r * 36) * 4;
    const float4 v = *(const float4*)(p.x + (size_t)(row0 + r) * 144 + c4);
    v4bf o;
    o[0] = (__bf16)v.x; o[1] = (__bf16)v.y;
    o[2] = (__bf16)v.z; o[3] = (__bf16)v.w;
    *(v4bf*)(rnnB + r * 288 + c4) = o;
  }
  {
    const int r = tid >> 4, c = tid & 15;  // 256 threads == 16*16 pad elems
    rnnB[r * 288 + 272 + c] = (__bf16)0.0f;
  }
  __syncthreads();

  // ---------------- decoder: 90 autoregressive steps ----------------
#pragma unroll 1
  for (int t = 0; t < 90; ++t) {
    gru_phase<9, 288>(rnnB, hB, U, p.PwihD, p.PwhhD, p.bihD, p.bhhD, wave, lane);
    __syncthreads();
    gate_phase(U, hS, hB, tid);
    __syncthreads();
    mlp2(hB, mB, p.PW1d, p.b1dp, p.PW2d, p.b2dp, tid, wave, lane);  // ends synced
    {  // output layer: 200 -> 144; write x_rec[t] and next step's x-part
      const __bf16* m2 = mB + 16 * 320;
      float* xrt = p.xrec + (size_t)t * (2048 * 144);
      for (int nt = wave; nt < 9; nt += 8) {
        v8f acc = gemm_bf<7, 224>(m2, p.PWo, nt, lane, vz);
        const int col = nt * 16 + nl;  // 9*16 == 144: always valid
        const float b = p.bo[col];
#pragma unroll
        for (int v = 0; v < 8; ++v) {
          const int rl = v + 8 * half;
          const float val = acc[v] + b;
          xrt[(size_t)(row0 + rl) * 144 + col] = val;
          rnnB[rl * 288 + col] = (__bf16)val;
        }
      }
    }
    __syncthreads();  // rnnB/U reuse ordering for next step
  }
}

// ---------------------------------------------------------------------------
// Weight packing: W (Ndim x Kdim row-major, B[k][n] = W[n][k]) -> WMMA B layout.
// n = nt*16 + (lane&15), k = kt*32 + 8*(lane>>4) + (j&7) + 16*(j>>3). Zero-padded, RNE.
// ---------------------------------------------------------------------------
__global__ __launch_bounds__(256) void pack_weight(
    const float* __restrict__ W, int Ndim, int Kdim, int KT,
    unsigned short* __restrict__ dst, int total) {
  const int i = blockIdx.x * 256 + threadIdx.x;
  if (i >= total) return;
  const int j = i & 15;
  const int lane = (i >> 4) & 31;
  const int tile = i >> 9;
  const int kt = tile % KT;
  const int nt = tile / KT;
  const int n = nt * 16 + (lane & 15);
  const int k = kt * 32 + 8 * (lane >> 4) + (j & 7) + 16 * (j >> 3);
  float v = (n < Ndim && k < Kdim) ? W[(size_t)n * Kdim + k] : 0.0f;
  unsigned int u = __float_as_uint(v);
  u += 0x7fffu + ((u >> 16) & 1u);  // round-to-nearest-even
  dst[i] = (unsigned short)(u >> 16);
}

__global__ __launch_bounds__(256) void pack_bias(const float* __restrict__ b,
                                                 int n, float* __restrict__ dst,
                                                 int npad) {
  const int i = blockIdx.x * 256 + threadIdx.x;
  if (i < npad) dst[i] = (i < n) ? b[i] : 0.0f;
}

// ---------------------------------------------------------------------------
// Host driver
// ---------------------------------------------------------------------------
extern "C" void kernel_launch(void* const* d_in, const int* in_sizes, int n_in,
                              void* d_out, int out_size, void* d_ws,
                              size_t ws_size, hipStream_t stream) {
  (void)in_sizes; (void)n_in; (void)out_size; (void)ws_size;
  constexpr int T = 90, B = 2048, NX = 144, NZ = 128;

  const float* x      = (const float*)d_in[0];
  const float* w_ih_e = (const float*)d_in[1];
  const float* w_hh_e = (const float*)d_in[2];
  const float* b_ih_e = (const float*)d_in[3];
  const float* b_hh_e = (const float*)d_in[4];
  const float* W1e    = (const float*)d_in[5];
  const float* b1e    = (const float*)d_in[6];
  const float* W2e    = (const float*)d_in[7];
  const float* b2e    = (const float*)d_in[8];
  const float* Wmu    = (const float*)d_in[9];
  const float* bmu    = (const float*)d_in[10];
  const float* Wlv    = (const float*)d_in[11];
  const float* blv    = (const float*)d_in[12];
  const float* w_ih_d = (const float*)d_in[13];
  const float* w_hh_d = (const float*)d_in[14];
  const float* b_ih_d = (const float*)d_in[15];
  const float* b_hh_d = (const float*)d_in[16];
  const float* W1d    = (const float*)d_in[17];
  const float* b1d    = (const float*)d_in[18];
  const float* W2d    = (const float*)d_in[19];
  const float* b2d    = (const float*)d_in[20];
  const float* Wo     = (const float*)d_in[21];
  const float* bo     = (const float*)d_in[22];

  float* xrec   = (float*)d_out;             // (T,B,NX)
  float* mu_out = xrec + (size_t)T * B * NX; // (B,NZ)
  float* lv_out = mu_out + (size_t)B * NZ;   // (B,NZ)

  // Workspace: padded biases (fp32), then packed bf16 weights.
  char* ws = (char*)d_ws;
  float* b1ep = (float*)ws;     // 304
  float* b2ep = b1ep + 304;     // 208
  float* b1dp = b2ep + 208;     // 304
  float* b2dp = b1dp + 304;     // 208
  size_t foff = (1024) * sizeof(float);
  unsigned short* pw = (unsigned short*)(ws + foff);
  unsigned short* PwihE = pw; pw += (size_t)24 * 5 * 512;
  unsigned short* PwhhE = pw; pw += (size_t)24 * 4 * 512;
  unsigned short* PwihD = pw; pw += (size_t)24 * 9 * 512;
  unsigned short* PwhhD = pw; pw += (size_t)24 * 4 * 512;
  unsigned short* PW1e  = pw; pw += (size_t)19 * 4 * 512;
  unsigned short* PW2e  = pw; pw += (size_t)13 * 10 * 512;
  unsigned short* PWmu  = pw; pw += (size_t)8 * 7 * 512;
  unsigned short* PWlv  = pw; pw += (size_t)8 * 7 * 512;
  unsigned short* PW1d  = pw; pw += (size_t)19 * 4 * 512;
  unsigned short* PW2d  = pw; pw += (size_t)13 * 10 * 512;
  unsigned short* PWo   = pw; pw += (size_t)9 * 7 * 512;

  auto pack = [&](const float* W, int N, int K, int KT, int NT,
                  unsigned short* dst) {
    const int total = NT * KT * 512;
    pack_weight<<<(total + 255) / 256, 256, 0, stream>>>(W, N, K, KT, dst, total);
  };
  pack(w_ih_e, 384, 144, 5, 24, PwihE);
  pack(w_hh_e, 384, 128, 4, 24, PwhhE);
  pack(w_ih_d, 384, 272, 9, 24, PwihD);
  pack(w_hh_d, 384, 128, 4, 24, PwhhD);
  pack(W1e, 300, 128, 4, 19, PW1e);
  pack(W2e, 200, 300, 10, 13, PW2e);
  pack(Wmu, 128, 200, 7, 8, PWmu);
  pack(Wlv, 128, 200, 7, 8, PWlv);
  pack(W1d, 300, 128, 4, 19, PW1d);
  pack(W2d, 200, 300, 10, 13, PW2d);
  pack(Wo, 144, 200, 7, 9, PWo);

  pack_bias<<<2, 256, 0, stream>>>(b1e, 300, b1ep, 304);
  pack_bias<<<1, 256, 0, stream>>>(b2e, 200, b2ep, 208);
  pack_bias<<<2, 256, 0, stream>>>(b1d, 300, b1dp, 304);
  pack_bias<<<1, 256, 0, stream>>>(b2d, 200, b2dp, 208);

  Params p;
  p.x = x; p.xrec = xrec; p.mu = mu_out; p.lv = lv_out;
  p.PwihE = PwihE; p.PwhhE = PwhhE; p.PwihD = PwihD; p.PwhhD = PwhhD;
  p.PW1e = PW1e; p.PW2e = PW2e; p.PWmu = PWmu; p.PWlv = PWlv;
  p.PW1d = PW1d; p.PW2d = PW2d; p.PWo = PWo;
  p.bihE = b_ih_e; p.bhhE = b_hh_e; p.bihD = b_ih_d; p.bhhD = b_hh_d;
  p.b1ep = b1ep; p.b2ep = b2ep; p.b1dp = b1dp; p.b2dp = b2dp;
  p.bmu = bmu; p.blv = blv; p.bo = bo;

  vae_all<<<B / 16, 256, 0, stream>>>(p);
}
